// MultiHeadAttention_27101243638248
// MI455X (gfx1250) — matrix operand order repaired and checked by reading, not compiled, not run on hardware
//
#include <hip/hip_runtime.h>
#include <hip/hip_bf16.h>
#include <math.h>

typedef __attribute__((ext_vector_type(16))) __bf16 v16bf;
typedef __attribute__((ext_vector_type(8)))  __bf16 v8bf;
typedef __attribute__((ext_vector_type(8)))  float  v8f;

#define D_MODEL     1024
#define N_HEADS     16
#define D_KDIM      64
#define N_IDX_HEADS 4
#define IDX_DIM     64
#define TOP_K_SEL   32
#define BB          2
#define TT          2048
#define EPSF        1.1920929e-07f

#define SHUF16(lo, hi) \
  __builtin_shufflevector(lo, hi, 0, 1, 2, 3, 4, 5, 6, 7, 8, 9, 10, 11, 12, 13, 14, 15)

// wave32 WMMA fragment maps (CDNA5 ISA 7.12.2), hf = lane>>4:
//  A (16x32 MxK): row = lane&15; elems = K[8hf..8hf+7] ++ K[16+8hf..16+8hf+7]
//  B (32x16 KxN): col = lane&15; elems = same K order as A   (both = 2 contiguous runs)
//  C/D (f32 16x16): col = lane&15; vgpr v -> row M = v + 8*hf

__device__ __forceinline__ float grp16_max(float v) {
#pragma unroll
  for (int off = 8; off >= 1; off >>= 1) v = fmaxf(v, __shfl_xor(v, off, 32));
  return v;
}
__device__ __forceinline__ float grp16_sum(float v) {
#pragma unroll
  for (int off = 8; off >= 1; off >>= 1) v += __shfl_xor(v, off, 32);
  return v;
}

// ---------------- elementwise f32 -> bf16 (counts are multiples of 4) -----------
__global__ __launch_bounds__(256) void cvt_f32_bf16(const float* __restrict__ s,
                                                    __bf16* __restrict__ d, long n) {
  long i = ((long)blockIdx.x * 256 + threadIdx.x) * 4;
  if (i + 3 >= n) return;
  float4 v = *(const float4*)(s + i);
  __bf16* o = d + i;
  o[0] = (__bf16)v.x; o[1] = (__bf16)v.y; o[2] = (__bf16)v.z; o[3] = (__bf16)v.w;
}

// ---------------- W[K][N] f32 -> Wt[Npad][K] bf16 (zero-pad cols) ---------------
__global__ __launch_bounds__(256) void wtrans(const float* __restrict__ W,
                                              __bf16* __restrict__ Wt,
                                              int K, int N, int Npad) {
  __shared__ float tile[32][33];
  const int k0 = blockIdx.x * 32, n0 = blockIdx.y * 32;
  const int tx = threadIdx.x, ty = threadIdx.y;  // (32,8)
  for (int yy = ty; yy < 32; yy += 8) {
    int k = k0 + yy, n = n0 + tx;
    tile[yy][tx] = (k < K && n < N) ? W[(size_t)k * N + n] : 0.0f;
  }
  __syncthreads();
  for (int yy = ty; yy < 32; yy += 8) {
    int n = n0 + yy, k = k0 + tx;
    if (n < Npad && k < K) Wt[(size_t)n * K + k] = (__bf16)tile[tx][yy];
  }
}

// ---------------- V[B,T,H,64] f32 -> Vt[B,H,64,T] bf16 -------------------------
__global__ __launch_bounds__(256) void vtrans(const float* __restrict__ V,
                                              __bf16* __restrict__ Vt) {
  __shared__ float tile[32][33];
  const int bh = blockIdx.z, b = bh >> 4, h = bh & 15;
  const int t0 = blockIdx.x * 32, d0 = blockIdx.y * 32;
  const int tx = threadIdx.x, ty = threadIdx.y;  // (32,8)
  for (int yy = ty; yy < 32; yy += 8)
    tile[yy][tx] = V[((size_t)(b * TT + t0 + yy) * N_HEADS + h) * D_KDIM + d0 + tx];
  __syncthreads();
  for (int yy = ty; yy < 32; yy += 8)
    Vt[((size_t)bh * D_KDIM + d0 + yy) * TT + t0 + tx] = (__bf16)tile[tx][yy];
}

// ---------------- GEMM: C[M,N] f32 = A[M,K]bf16 @ Bt[Npad,K]bf16 ----------------
// one wave per 32(M) x 64(N) tile; ping-pong double-buffered fragments (no copies);
// 8 WMMAs per K-chunk fed by 12 b128 loads; requires K % 64 == 0, K >= 64
__global__ __launch_bounds__(32) void gemm_bf16_wmma(
    const __bf16* __restrict__ A, const __bf16* __restrict__ Bt,
    float* __restrict__ C, int M, int N, int K) {
  const int lane = threadIdx.x, hf = lane >> 4, nn = lane & 15;
  const int m0 = blockIdx.y * 32, n0 = blockIdx.x * 64;
  const __bf16* arow0 = A + (size_t)(m0 + nn) * K + 8 * hf;
  const __bf16* arow1 = A + (size_t)(m0 + 16 + nn) * K + 8 * hf;
  const __bf16* brow[4];
#pragma unroll
  for (int j = 0; j < 4; ++j)
    brow[j] = Bt + (size_t)(n0 + j * 16 + nn) * K + 8 * hf;

  v8f acc[8];
#pragma unroll
  for (int j = 0; j < 8; ++j)
#pragma unroll
    for (int i = 0; i < 8; ++i) acc[j][i] = 0.0f;

#define LDA(r, kk) SHUF16(*(const v8bf*)((r) + (kk)), *(const v8bf*)((r) + (kk) + 16))
#define LDB(j, kk) \
  SHUF16(*(const v8bf*)(brow[j] + (kk)), *(const v8bf*)(brow[j] + (kk) + 16))

  auto compute = [&](const v16bf& ca0, const v16bf& ca1, const v16bf (&cb)[4]) {
#pragma unroll
    for (int j = 0; j < 4; ++j) {
      acc[j] = __builtin_amdgcn_wmma_f32_16x16x32_bf16(
          false, ca0, false, cb[j], (short)0, acc[j], false, false);
      acc[4 + j] = __builtin_amdgcn_wmma_f32_16x16x32_bf16(
          false, ca1, false, cb[j], (short)0, acc[4 + j], false, false);
    }
  };

  // ping-pong buffers: buf0 = even chunks, buf1 = odd chunks
  v16bf a0_0 = LDA(arow0, 0), a1_0 = LDA(arow1, 0);
  v16bf b_0[4];
#pragma unroll
  for (int j = 0; j < 4; ++j) b_0[j] = LDB(j, 0);
  v16bf a0_1 = LDA(arow0, 32), a1_1 = LDA(arow1, 32);
  v16bf b_1[4];
#pragma unroll
  for (int j = 0; j < 4; ++j) b_1[j] = LDB(j, 32);

  for (int kk = 0; kk < K - 64; kk += 64) {
    __builtin_prefetch(arow0 + kk + 512, 0, 1);
    __builtin_prefetch(brow[0] + kk + 512, 0, 1);
    compute(a0_0, a1_0, b_0);  // chunk kk; overlaps buf1 loads from prev iter
    a0_0 = LDA(arow0, kk + 64);
    a1_0 = LDA(arow1, kk + 64);
#pragma unroll
    for (int j = 0; j < 4; ++j) b_0[j] = LDB(j, kk + 64);
    compute(a0_1, a1_1, b_1);  // chunk kk+32; overlaps buf0 loads above
    a0_1 = LDA(arow0, kk + 96);
    a1_1 = LDA(arow1, kk + 96);
#pragma unroll
    for (int j = 0; j < 4; ++j) b_1[j] = LDB(j, kk + 96);
  }
  compute(a0_0, a1_0, b_0);  // chunk K-64
  compute(a0_1, a1_1, b_1);  // chunk K-32
#undef LDA
#undef LDB

#pragma unroll
  for (int mt = 0; mt < 2; ++mt)
#pragma unroll
    for (int j = 0; j < 4; ++j)
#pragma unroll
      for (int v = 0; v < 8; ++v) {
        int gm = m0 + mt * 16 + v + 8 * hf;
        int gn = n0 + j * 16 + nn;
        if (gm < M && gn < N) C[(size_t)gm * N + gn] = acc[mt * 4 + j][v];
      }
}

// ---------------- fused RMSNorm + RoPE, f32 in -> bf16 out ----------------------
__global__ __launch_bounds__(32) void norm_rope(const float* __restrict__ X,
                                                const float* __restrict__ g,
                                                __bf16* __restrict__ Y, int heads) {
  const int blk = blockIdx.x;  // (b*T + t)*heads + h
  const int t = (blk / heads) % TT;
  const int lane = threadIdx.x;  // pair index
  const size_t base = (size_t)blk * 64;
  float x0 = X[base + 2 * lane], x1 = X[base + 2 * lane + 1];
  float ss = x0 * x0 + x1 * x1;
#pragma unroll
  for (int off = 16; off >= 1; off >>= 1) ss += __shfl_xor(ss, off, 32);
  float r = rsqrtf(ss * (1.0f / 64.0f) + EPSF);
  float y0 = x0 * r * g[2 * lane];
  float y1 = x1 * r * g[2 * lane + 1];
  float inv = powf(10000.0f, -((float)(2 * lane)) / 64.0f);
  float ang = (float)t * inv;
  float c = cosf(ang), s = sinf(ang);
  Y[base + 2 * lane]     = (__bf16)(y0 * c - y1 * s);
  Y[base + 2 * lane + 1] = (__bf16)(y1 * c + y0 * s);
}

// ---------------- indexer scores (all fragments direct b128 loads) --------------
__global__ __launch_bounds__(32) void indexer_scores(
    const __bf16* __restrict__ qI, const __bf16* __restrict__ kI,
    const float* __restrict__ wI, float* __restrict__ ISC) {
  const int k0 = blockIdx.x * 16, q0 = blockIdx.y * 16, b = blockIdx.z;
  if (k0 > q0 + 15) return;  // fully non-causal tile: never read downstream
  const int lane = threadIdx.x, hf = lane >> 4, nn = lane & 15;
  const __bf16* krow = kI + (size_t)(b * TT + k0 + nn) * IDX_DIM + 8 * hf;
  float score[8];
#pragma unroll
  for (int v = 0; v < 8; ++v) score[v] = 0.0f;
#pragma unroll
  for (int h = 0; h < N_IDX_HEADS; ++h) {
    const __bf16* qrow =
        qI + ((size_t)(b * TT + q0 + nn) * N_IDX_HEADS + h) * IDX_DIM;
    v8f acc;
#pragma unroll
    for (int i = 0; i < 8; ++i) acc[i] = 0.0f;
#pragma unroll
    for (int cc = 0; cc < IDX_DIM; cc += 32) {
      v16bf a = SHUF16(*(const v8bf*)(qrow + cc + 8 * hf),
                       *(const v8bf*)(qrow + cc + 16 + 8 * hf));
      v16bf bb = SHUF16(*(const v8bf*)(krow + cc),
                        *(const v8bf*)(krow + cc + 16));
      acc = __builtin_amdgcn_wmma_f32_16x16x32_bf16(
          false, a, false, bb, (short)0, acc, false, false);
    }
#pragma unroll
    for (int v = 0; v < 8; ++v) {
      int q = q0 + v + 8 * hf;
      score[v] += fmaxf(acc[v], 0.0f) * wI[(size_t)(b * TT + q) * N_IDX_HEADS + h];
    }
  }
#pragma unroll
  for (int v = 0; v < 8; ++v) {
    int q = q0 + v + 8 * hf;
    ISC[(size_t)(b * TT + q) * TT + k0 + nn] = score[v];
  }
}

// ---------------- per-row 32nd-largest causal score -----------------------------
__global__ __launch_bounds__(32) void topk_threshold(const float* __restrict__ ISC,
                                                     float* __restrict__ THR) {
  const int row = blockIdx.x;  // b*T + q
  const int q = row % TT;
  const int lane = threadIdx.x;
  const int n = q + 1;
  if (n <= TOP_K_SEL) {  // top-32 includes -inf entries -> everything passes
    if (lane == 0) THR[row] = -__builtin_inff();
    return;
  }
  const float* r = ISC + (size_t)row * TT;
  float prev = __builtin_inff();
  for (int it = 0; it < TOP_K_SEL; ++it) {
    float local = -__builtin_inff();
    for (int k = lane; k < n; k += 32) {
      float v = r[k];
      if (v < prev) local = fmaxf(local, v);
    }
#pragma unroll
    for (int off = 16; off >= 1; off >>= 1)
      local = fmaxf(local, __shfl_xor(local, off, 32));
    prev = local;
  }
  if (lane == 0) THR[row] = prev;
}

// ---------------- flash attention with top-k mask, bf16 out ----------------------
__global__ __launch_bounds__(32) void flash_attn(
    const __bf16* __restrict__ Qh, const __bf16* __restrict__ Kh,
    const __bf16* __restrict__ Vt, const float* __restrict__ ISC,
    const float* __restrict__ THR, __bf16* __restrict__ O) {
  __shared__ __bf16 Ps[16][32];  // P: C-layout -> A-layout relay (vector ds ops)
  const int q0 = blockIdx.x * 16, h = blockIdx.y, b = blockIdx.z;
  const int lane = threadIdx.x, hf = lane >> 4, nn = lane & 15;
  const float scale = 0.125f;  // 1/sqrt(64)

  const __bf16* qrow = Qh + ((size_t)(b * TT + q0 + nn) * N_HEADS + h) * D_KDIM;
  v16bf aq0 = SHUF16(*(const v8bf*)(qrow + 8 * hf),
                     *(const v8bf*)(qrow + 16 + 8 * hf));
  v16bf aq1 = SHUF16(*(const v8bf*)(qrow + 32 + 8 * hf),
                     *(const v8bf*)(qrow + 48 + 8 * hf));

  float tv[8], mrow[8], lrow[8];
  v8f o[4];
#pragma unroll
  for (int v = 0; v < 8; ++v) {
    int q = q0 + v + 8 * hf;
    tv[v] = THR[(size_t)b * TT + q];
    mrow[v] = -__builtin_inff();
    lrow[v] = 0.0f;
  }
#pragma unroll
  for (int t = 0; t < 4; ++t)
#pragma unroll
    for (int i = 0; i < 8; ++i) o[t][i] = 0.0f;

  const int kend = q0 + 15;
  for (int k0 = 0; k0 <= kend; k0 += 32) {
    float sreg[2][8];
#pragma unroll
    for (int j = 0; j < 2; ++j) {
      const __bf16* krow =
          Kh + ((size_t)(b * TT + k0 + j * 16 + nn) * N_HEADS + h) * D_KDIM;
      v16bf bk0 = SHUF16(*(const v8bf*)(krow + 8 * hf),
                         *(const v8bf*)(krow + 16 + 8 * hf));
      v16bf bk1 = SHUF16(*(const v8bf*)(krow + 32 + 8 * hf),
                         *(const v8bf*)(krow + 48 + 8 * hf));
      v8f sa;
#pragma unroll
      for (int i = 0; i < 8; ++i) sa[i] = 0.0f;
      sa = __builtin_amdgcn_wmma_f32_16x16x32_bf16(false, aq0, false, bk0,
                                                   (short)0, sa, false, false);
      sa = __builtin_amdgcn_wmma_f32_16x16x32_bf16(false, aq1, false, bk1,
                                                   (short)0, sa, false, false);
#pragma unroll
      for (int v = 0; v < 8; ++v) {
        int q = q0 + v + 8 * hf;
        int key = k0 + j * 16 + nn;
        bool ok = (key <= q) && (ISC[(size_t)(b * TT + q) * TT + key] >= tv[v]);
        sreg[j][v] = sa[v] * scale + (ok ? 0.0f : -1.0e9f);
      }
    }
    // online softmax per row (row M = v + 8*hf lives in one 16-lane group)
#pragma unroll
    for (int v = 0; v < 8; ++v) {
      float mx = fmaxf(grp16_max(sreg[0][v]), grp16_max(sreg[1][v]));
      float nm = fmaxf(mrow[v], mx);
      float corr = __expf(mrow[v] - nm);
      mrow[v] = nm;
      float p0 = __expf(sreg[0][v] - nm);
      float p1 = __expf(sreg[1][v] - nm);
      lrow[v] = lrow[v] * corr + grp16_sum(p0 + p1);
#pragma unroll
      for (int t = 0; t < 4; ++t) o[t][v] *= corr;
      Ps[v + 8 * hf][nn] = (__bf16)p0;
      Ps[v + 8 * hf][16 + nn] = (__bf16)p1;
    }
    v16bf pa = SHUF16(*(const v8bf*)&Ps[nn][16 * hf],
                      *(const v8bf*)&Ps[nn][16 * hf + 8]);
#pragma unroll
    for (int t = 0; t < 4; ++t) {
      const __bf16* vrow =
          Vt + ((size_t)(b * N_HEADS + h) * D_KDIM + t * 16 + nn) * TT + k0 + 16 * hf;
      v16bf bv = SHUF16(*(const v8bf*)vrow, *(const v8bf*)(vrow + 8));
      o[t] = __builtin_amdgcn_wmma_f32_16x16x32_bf16(false, pa, false, bv,
                                                     (short)0, o[t], false, false);
    }
  }
#pragma unroll
  for (int t = 0; t < 4; ++t)
#pragma unroll
    for (int v = 0; v < 8; ++v) {
      int q = q0 + v + 8 * hf;
      O[(size_t)(b * TT + q) * D_MODEL + h * D_KDIM + t * 16 + nn] =
          (__bf16)(o[t][v] / lrow[v]);
    }
}

extern "C" void kernel_launch(void* const* d_in, const int* in_sizes, int n_in,
                              void* d_out, int out_size, void* d_ws, size_t ws_size,
                              hipStream_t stream) {
  const float* x   = (const float*)d_in[0];
  const float* Wq  = (const float*)d_in[1];
  const float* Wk  = (const float*)d_in[2];
  const float* Wv  = (const float*)d_in[3];
  const float* Wo  = (const float*)d_in[4];
  const float* gq  = (const float*)d_in[5];
  const float* gk  = (const float*)d_in[6];
  const float* Wiq = (const float*)d_in[7];
  const float* Wiw = (const float*)d_in[8];
  const float* Wik = (const float*)d_in[9];
  float* out = (float*)d_out;

  const int BT = BB * TT;  // 4096
  char* p = (char*)d_ws;
  auto alloc = [&](size_t bytes) {
    void* r = (void*)p;
    p += (bytes + 255) & ~(size_t)255;
    return r;
  };
  // f32 intermediates
  float* Qf  = (float*)alloc((size_t)BT * D_MODEL * 4);
  float* Kf  = (float*)alloc((size_t)BT * D_MODEL * 4);
  float* Vf  = (float*)alloc((size_t)BT * D_MODEL * 4);
  float* qIf = (float*)alloc((size_t)BT * N_IDX_HEADS * IDX_DIM * 4);
  float* kIf = (float*)alloc((size_t)BT * IDX_DIM * 4);
  float* wI  = (float*)alloc((size_t)BT * N_IDX_HEADS * 4);
  float* ISC = (float*)alloc((size_t)BT * TT * 4);
  float* THR = (float*)alloc((size_t)BT * 4);
  // bf16 operands
  __bf16* xh   = (__bf16*)alloc((size_t)BT * D_MODEL * 2);
  __bf16* Wqt  = (__bf16*)alloc((size_t)D_MODEL * D_MODEL * 2);
  __bf16* Wkt  = (__bf16*)alloc((size_t)D_MODEL * D_MODEL * 2);
  __bf16* Wvt  = (__bf16*)alloc((size_t)D_MODEL * D_MODEL * 2);
  __bf16* Wot  = (__bf16*)alloc((size_t)D_MODEL * D_MODEL * 2);
  __bf16* Wiqt = (__bf16*)alloc((size_t)(N_IDX_HEADS * IDX_DIM) * D_MODEL * 2);
  __bf16* Wikt = (__bf16*)alloc((size_t)64 * D_MODEL * 2);
  __bf16* Wiwt = (__bf16*)alloc((size_t)64 * D_MODEL * 2);  // N=4 padded to 64
  __bf16* Qh   = (__bf16*)alloc((size_t)BT * D_MODEL * 2);
  __bf16* Kh   = (__bf16*)alloc((size_t)BT * D_MODEL * 2);
  __bf16* Vth  = (__bf16*)alloc((size_t)BT * D_MODEL * 2);  // [B,H,64,T]
  __bf16* qIh  = (__bf16*)alloc((size_t)BT * N_IDX_HEADS * IDX_DIM * 2);
  __bf16* kIh  = (__bf16*)alloc((size_t)BT * IDX_DIM * 2);
  __bf16* AOh  = (__bf16*)alloc((size_t)BT * D_MODEL * 2);

  dim3 tb(32, 8);
  auto ggrid = [](int M, int N) { return dim3((N + 63) / 64, M / 32); };
  auto tgrid = [](int K, int Npad) { return dim3((K + 31) / 32, (Npad + 31) / 32); };

  // operand prep
  cvt_f32_bf16<<<(BT * D_MODEL) / 1024, 256, 0, stream>>>(x, xh, (long)BT * D_MODEL);
  wtrans<<<tgrid(D_MODEL, D_MODEL), tb, 0, stream>>>(Wq, Wqt, D_MODEL, D_MODEL, D_MODEL);
  wtrans<<<tgrid(D_MODEL, D_MODEL), tb, 0, stream>>>(Wk, Wkt, D_MODEL, D_MODEL, D_MODEL);
  wtrans<<<tgrid(D_MODEL, D_MODEL), tb, 0, stream>>>(Wv, Wvt, D_MODEL, D_MODEL, D_MODEL);
  wtrans<<<tgrid(D_MODEL, D_MODEL), tb, 0, stream>>>(Wo, Wot, D_MODEL, D_MODEL, D_MODEL);
  wtrans<<<tgrid(D_MODEL, 256), tb, 0, stream>>>(Wiq, Wiqt, D_MODEL, 256, 256);
  wtrans<<<tgrid(D_MODEL, 64), tb, 0, stream>>>(Wik, Wikt, D_MODEL, 64, 64);
  wtrans<<<tgrid(D_MODEL, 64), tb, 0, stream>>>(Wiw, Wiwt, D_MODEL, N_IDX_HEADS, 64);

  // projections
  gemm_bf16_wmma<<<ggrid(BT, D_MODEL), 32, 0, stream>>>(xh, Wqt, Qf, BT, D_MODEL, D_MODEL);
  gemm_bf16_wmma<<<ggrid(BT, D_MODEL), 32, 0, stream>>>(xh, Wkt, Kf, BT, D_MODEL, D_MODEL);
  gemm_bf16_wmma<<<ggrid(BT, D_MODEL), 32, 0, stream>>>(xh, Wvt, Vf, BT, D_MODEL, D_MODEL);
  gemm_bf16_wmma<<<ggrid(BT, 256), 32, 0, stream>>>(xh, Wiqt, qIf, BT, 256, D_MODEL);
  gemm_bf16_wmma<<<ggrid(BT, 64), 32, 0, stream>>>(xh, Wikt, kIf, BT, 64, D_MODEL);
  gemm_bf16_wmma<<<ggrid(BT, 64), 32, 0, stream>>>(xh, Wiwt, wI, BT, N_IDX_HEADS, D_MODEL);

  // rmsnorm + rope -> bf16; V transpose -> bf16; indexer operands -> bf16
  norm_rope<<<BT * N_HEADS, 32, 0, stream>>>(Qf, gq, Qh, N_HEADS);
  norm_rope<<<BT * N_HEADS, 32, 0, stream>>>(Kf, gk, Kh, N_HEADS);
  vtrans<<<dim3(TT / 32, 2, BB * N_HEADS), tb, 0, stream>>>(Vf, Vth);
  cvt_f32_bf16<<<(BT * N_IDX_HEADS * IDX_DIM) / 1024, 256, 0, stream>>>(
      qIf, qIh, (long)BT * N_IDX_HEADS * IDX_DIM);
  cvt_f32_bf16<<<(BT * IDX_DIM) / 1024, 256, 0, stream>>>(kIf, kIh, (long)BT * IDX_DIM);

  // indexer scores + per-row top-k threshold
  indexer_scores<<<dim3(TT / 16, TT / 16, BB), 32, 0, stream>>>(qIh, kIh, wI, ISC);
  topk_threshold<<<BT, 32, 0, stream>>>(ISC, THR);

  // sparse-masked flash attention (bf16 out feeds final GEMM directly)
  flash_attn<<<dim3(TT / 16, N_HEADS, BB), 32, 0, stream>>>(Qh, Kh, Vth, ISC, THR, AOh);

  // output projection
  gemm_bf16_wmma<<<ggrid(BT, D_MODEL), 32, 0, stream>>>(AOh, Wot, out, BT, D_MODEL, D_MODEL);
}
